// RNN_89610197664263
// MI455X (gfx1250) — matrix operand order repaired and checked by reading, not compiled, not run on hardware
//
#include <hip/hip_runtime.h>

typedef __attribute__((ext_vector_type(16))) _Float16 v16h;
typedef __attribute__((ext_vector_type(8)))  _Float16 v8h;
typedef __attribute__((ext_vector_type(8)))  float    v8f;
typedef __attribute__((ext_vector_type(4)))  float    f4;

#define T_LEN   28
#define EMBED   28
#define HID     256
#define OUT_N   10
#define BM      128          // batch rows per block
#define WAVES   8
#define MROWS   16           // rows per wave

// LDS layout (padded strides avoid 64-bank conflicts on b128 reads)
#define W2_STRIDE 264        // halves per W2 row   (264*2 B = 528 B -> lanes spread over all banks)
#define W1_STRIDE 40         // halves per W1 row   (80 B stride)
#define H_STRIDE  264        // halves per h row

#define OFF_W2   0
#define OFF_W1   (OFF_W2 + 256*W2_STRIDE*2)              // 135168
#define OFF_W3   (OFF_W1 + 256*W1_STRIDE*2)              // +20480
#define OFF_B12  (OFF_W3 + 16*W2_STRIDE*2)               // +8448
#define OFF_B3   (OFF_B12 + 256*4)                       // +1024
#define OFF_H    (OFF_B3  + 16*4)                        // +64
#define LDS_TOTAL (OFF_H + WAVES*MROWS*H_STRIDE*2)       // = 232768 B  (< 320 KB/WGP)

__global__ __launch_bounds__(256) void rnn_fused_kernel(
    const float* __restrict__ seq, const float* __restrict__ W1,
    const float* __restrict__ b1,  const float* __restrict__ W2,
    const float* __restrict__ b2,  const float* __restrict__ W3,
    const float* __restrict__ b3,  float* __restrict__ out)
{
    extern __shared__ char smem[];
    _Float16* w2s = (_Float16*)(smem + OFF_W2);
    _Float16* w1s = (_Float16*)(smem + OFF_W1);
    _Float16* w3s = (_Float16*)(smem + OFF_W3);
    float*    b12 = (float*)   (smem + OFF_B12);
    float*    b3s = (float*)   (smem + OFF_B3);
    _Float16* hs  = (_Float16*)(smem + OFF_H);

    const int tid = threadIdx.x;

    // ---------------- init LDS: f16 weights (+K/N padding), fused biases, h=0 ----------------
    for (int i = tid; i < HID*HID; i += 256) {
        int r = i >> 8, c = i & 255;
        w2s[r*W2_STRIDE + c] = (_Float16)W2[i];
    }
    for (int i = tid; i < HID*32; i += 256) {
        int r = i >> 5, c = i & 31;
        w1s[r*W1_STRIDE + c] = (c < EMBED) ? (_Float16)W1[r*EMBED + c] : (_Float16)0.f;
    }
    for (int i = tid; i < 16*HID; i += 256) {
        int r = i >> 8, c = i & 255;
        w3s[r*W2_STRIDE + c] = (r < OUT_N) ? (_Float16)W3[r*HID + c] : (_Float16)0.f;
    }
    if (tid < HID) b12[tid] = b1[tid] + b2[tid];
    if (tid < 16)  b3s[tid] = (tid < OUT_N) ? b3[tid] : 0.f;
    for (int i = tid; i < WAVES*MROWS*H_STRIDE; i += 256) hs[i] = (_Float16)0.f;
    __syncthreads();

    const int lane = tid & 31;
    const int wv   = tid >> 5;
    const int mr   = lane & 15;   // A: row-in-tile ; B: col-in-tile ; D: n index
    const int half = lane >> 4;   // K-half selector per ISA fragment layouts

    _Float16* hrow  = hs + (wv*MROWS + mr) * H_STRIDE;   // this lane's A-matrix row of h
    _Float16* hwave = hs + wv*MROWS*H_STRIDE;            // wave-private h block
    const long bglob = (long)blockIdx.x * BM + wv*MROWS + mr;

    // ---------------- time loop (no inter-wave sync needed: h regions are wave-private) ------
    for (int t = 0; t < T_LEN; ++t) {
        // A fragment of seq[:, t, :] (K = EMBED padded to 32), f32 global -> f16.
        // Row base (b*28+t)*28 floats is 16B aligned (28 % 4 == 0).
        union { v16h v; _Float16 e[16]; } aseq;
        const float* srow = seq + (bglob * T_LEN + t) * EMBED;
        {
            const int e0 = half * 8;        // K = e0..e0+7      (always < 28)
            f4 x0 = *(const f4*)(srow + e0);
            f4 x1 = *(const f4*)(srow + e0 + 4);
            const int e1 = 16 + half * 8;   // K = e1..e1+7      (28..31 must be zero)
            f4 y0 = *(const f4*)(srow + e1);
            f4 y1 = {};
            if (half == 0) y1 = *(const f4*)(srow + e1 + 4);
            #pragma unroll
            for (int j = 0; j < 4; ++j) {
                aseq.e[j]      = (_Float16)x0[j];
                aseq.e[4 + j]  = (_Float16)x1[j];
                aseq.e[8 + j]  = (_Float16)y0[j];
                aseq.e[12 + j] = (_Float16)y1[j];
            }
        }
        // A fragments of h strip (K = 256 -> 8 fragments), from wave-private LDS
        v16h ah[8];
        #pragma unroll
        for (int kk = 0; kk < 8; ++kk) {
            const int kb = kk * 32;
            union { v16h v; v8h h2[2]; } u;
            u.h2[0] = *(const v8h*)(hrow + kb + half*8);        // K = kb + half*8 .. +7
            u.h2[1] = *(const v8h*)(hrow + kb + 16 + half*8);   // K = kb+16+half*8 .. +7
            ah[kk] = u.v;
        }

        for (int nt = 0; nt < 16; ++nt) {
            v8f acc = {};
            // input projection: one K=32 WMMA against zero-padded W1
            {
                union { v16h v; v8h h2[2]; } bf;
                const _Float16* p = w1s + (nt*16 + mr)*W1_STRIDE + half*8;
                bf.h2[0] = *(const v8h*)(p);
                bf.h2[1] = *(const v8h*)(p + 16);
                acc = __builtin_amdgcn_wmma_f32_16x16x32_f16(
                        false, aseq.v, false, bf.v, (short)0, acc, false, false);
            }
            // recurrence: h @ W2^T, 8 K-steps
            #pragma unroll
            for (int kk = 0; kk < 8; ++kk) {
                union { v16h v; v8h h2[2]; } bf;
                const _Float16* p = w2s + (nt*16 + mr)*W2_STRIDE + kk*32 + half*8;
                bf.h2[0] = *(const v8h*)(p);
                bf.h2[1] = *(const v8h*)(p + 16);
                acc = __builtin_amdgcn_wmma_f32_16x16x32_f16(
                        false, ah[kk], false, bf.v, (short)0, acc, false, false);
            }
            // bias (n = mr is constant per lane), tanh via v_exp, write back f16 h
            const float bv = b12[nt*16 + mr];
            #pragma unroll
            for (int v = 0; v < 8; ++v) {
                float x  = acc[v] + bv;
                float ex = __expf(2.f * x);
                float th = 1.f - 2.f / (ex + 1.f);     // == tanh(x), correct limits at +/-inf
                const int m = v + 8*half;              // D layout: vgpr v, lane half -> M
                hwave[m*H_STRIDE + nt*16 + mr] = (_Float16)th;
            }
        }
        // same-wave DS ops are in-order; next iteration's reads see these writes
    }

    // ---------------- output head: h @ W3^T + b3 (N padded 10 -> 16) -------------------------
    {
        v16h ah[8];
        #pragma unroll
        for (int kk = 0; kk < 8; ++kk) {
            const int kb = kk * 32;
            union { v16h v; v8h h2[2]; } u;
            u.h2[0] = *(const v8h*)(hrow + kb + half*8);
            u.h2[1] = *(const v8h*)(hrow + kb + 16 + half*8);
            ah[kk] = u.v;
        }
        v8f acc = {};
        #pragma unroll
        for (int kk = 0; kk < 8; ++kk) {
            union { v16h v; v8h h2[2]; } bf;
            const _Float16* p = w3s + mr*W2_STRIDE + kk*32 + half*8;
            bf.h2[0] = *(const v8h*)(p);
            bf.h2[1] = *(const v8h*)(p + 16);
            acc = __builtin_amdgcn_wmma_f32_16x16x32_f16(
                    false, ah[kk], false, bf.v, (short)0, acc, false, false);
        }
        if (mr < OUT_N) {
            const float bo = b3s[mr];
            #pragma unroll
            for (int v = 0; v < 8; ++v) {
                const int m = v + 8*half;
                const long row = (long)blockIdx.x * BM + wv*MROWS + m;
                out[row*OUT_N + mr] = acc[v] + bo;
            }
        }
    }
}

extern "C" void kernel_launch(void* const* d_in, const int* in_sizes, int n_in,
                              void* d_out, int out_size, void* d_ws, size_t ws_size,
                              hipStream_t stream) {
    const float* seq = (const float*)d_in[0];
    const float* W1  = (const float*)d_in[1];
    const float* b1  = (const float*)d_in[2];
    const float* W2  = (const float*)d_in[3];
    const float* b2  = (const float*)d_in[4];
    const float* W3  = (const float*)d_in[5];
    const float* b3  = (const float*)d_in[6];
    float* out = (float*)d_out;

    const int B = in_sizes[0] / (T_LEN * EMBED);   // 32768
    const int grid = B / BM;                       // 256 blocks

    (void)hipFuncSetAttribute((const void*)rnn_fused_kernel,
                              hipFuncAttributeMaxDynamicSharedMemorySize, LDS_TOTAL);
    rnn_fused_kernel<<<grid, 256, LDS_TOTAL, stream>>>(seq, W1, b1, W2, b2, W3, b3, out);
}
